// MultiHeadAttention_34213709480554
// MI455X (gfx1250) — matrix operand order repaired and checked by reading, not compiled, not run on hardware
//
#include <hip/hip_runtime.h>

// ---------------------------------------------------------------------------
// MHA for MI455X (gfx1250): all GEMMs via v_wmma_f32_16x16x32_bf16 (wave32).
//  k0 : elementwise fp32->bf16 pre-pass for x, w_qkv, w_proj (hoists all
//       dtype conversion out of the GEMM hot loops)
//  k1 : QKV GEMM (bf16 x bf16 -> f32 acc), epilogue scatters bf16 Q (scaled),
//       K, and V^T
//  k2 : attention: S = Q K^T into 64 KB bf16 LDS, in-place vectorized softmax,
//       O = P V with pure ds_load_b128 fragments (no conversion in loop)
//  k3 : proj GEMM (bf16 x bf16), fp32 out + bias
// Workspace: Xbf 16M + Wqkv 6M + Wproj 2M + Q/K/VT/AO 4x16M = 88 MiB.
// ---------------------------------------------------------------------------

typedef __attribute__((ext_vector_type(16))) __bf16          v16bf;
typedef __attribute__((ext_vector_type(8)))  float           v8f;
typedef __attribute__((ext_vector_type(4)))  float           f32x4;
typedef __attribute__((ext_vector_type(8)))  unsigned short  u16x8;
typedef __attribute__((ext_vector_type(16))) unsigned short  u16x16;
typedef unsigned short u16;

#define EMB    1024
#define NSEQ   2048
#define NBATCH 4
#define NHEAD  16
#define HDIM   64
#define MTOT   (NBATCH * NSEQ)   // 8192
#define NQKV   (3 * EMB)         // 3072

static __device__ __forceinline__ u16 f2bf(float f) {
    unsigned u = __builtin_bit_cast(unsigned, f);
    u += 0x7FFFu + ((u >> 16) & 1u);          // round-to-nearest-even (approx)
    return (u16)(u >> 16);
}

static __device__ __forceinline__ float bf2f(u16 v) {
    unsigned u = ((unsigned)v) << 16;
    return __builtin_bit_cast(float, u);
}

static __device__ __forceinline__ v8f wmma_bf16(v16bf a, v16bf b, v8f c) {
    // (neg_a, A, neg_b, B, c_mod, C, reuse_a, reuse_b)
    return __builtin_amdgcn_wmma_f32_16x16x32_bf16(false, a, false, b, (short)0, c,
                                                   false, false);
}

// A fragment (16x32 bf16) from row-major bf16 src, leading dim ld.
// ISA layout: lanes 0-15 row M=l, K={kb..kb+7, kb+16..kb+23} with kb=0;
// lanes 16-31 same rows with kb=8.  Two 16-byte loads per lane.
static __device__ __forceinline__ v16bf load_a_bf16(const u16* __restrict__ src, int ld,
                                                    int m0, int k0, int lane) {
    int row = lane & 15;
    int kb  = (lane >> 4) << 3;
    const u16* p = src + (size_t)(m0 + row) * ld + k0 + kb;
    u16x8 lo = *(const u16x8*)(p);        // K = kb+0..7
    u16x8 hi = *(const u16x8*)(p + 16);   // K = kb+16..23
    u16x16 u;
#pragma unroll
    for (int i = 0; i < 8; ++i) { u[i] = lo[i]; u[i + 8] = hi[i]; }
    return __builtin_bit_cast(v16bf, u);
}

// B fragment (32x16 bf16), element (k, n) at src[n*ld + k] (K-contiguous).
// Lane holds column n = l&15; elems e<8 -> K = kh + e, e>=8 -> K = kh + 8 + e, kh = (l>=16)*8.
static __device__ __forceinline__ v16bf load_b_bf16(const u16* __restrict__ src, int ld,
                                                    int k0, int n0, int lane) {
    int col = lane & 15;
    int kh  = (lane >> 4) << 3;
    const u16* p = src + (size_t)(n0 + col) * ld + k0 + kh;
    u16x8 lo = *(const u16x8*)(p);        // K = kh+0..7
    u16x8 hi = *(const u16x8*)(p + 16);   // K = kh+16..23
    u16x16 u;
#pragma unroll
    for (int i = 0; i < 8; ++i) { u[i] = lo[i]; u[i + 8] = hi[i]; }
    return __builtin_bit_cast(v16bf, u);
}

// ---------------------------------------------------------------------------
// Kernel 0: elementwise fp32 -> bf16 (8 elems/thread, b128 in, b128 out)
// ---------------------------------------------------------------------------
__global__ __launch_bounds__(256) void cvt_f32_bf16(const float* __restrict__ in,
                                                    u16* __restrict__ outp)
{
    size_t i = ((size_t)blockIdx.x * blockDim.x + threadIdx.x) * 8;
    f32x4 a = *(const f32x4*)(in + i);
    f32x4 b = *(const f32x4*)(in + i + 4);
    u16x8 o;
#pragma unroll
    for (int j = 0; j < 4; ++j) { o[j] = f2bf(a[j]); o[j + 4] = f2bf(b[j]); }
    *(u16x8*)(outp + i) = o;
}

// ---------------------------------------------------------------------------
// Kernel 1: qkv = x @ w_qkv^T + b_qkv, scattered to Q (scaled), K, V^T (bf16).
// Block = 128 thr (4 waves, 2x2), tile 128x128; wave tile 64x64 (4x4 WMMA).
// ---------------------------------------------------------------------------
__global__ __launch_bounds__(128) void qkv_gemm_scatter(
    const u16* __restrict__ x, const u16* __restrict__ w,
    const float* __restrict__ bias,
    u16* __restrict__ Qo, u16* __restrict__ Ko, u16* __restrict__ VTo)
{
    const int lane = threadIdx.x & 31;
    const int wave = threadIdx.x >> 5;
    const int m0w  = blockIdx.y * 128 + (wave >> 1) * 64;
    const int n0w  = blockIdx.x * 128 + (wave & 1) * 64;

    v8f acc[4][4];
#pragma unroll
    for (int i = 0; i < 4; ++i)
#pragma unroll
        for (int j = 0; j < 4; ++j) acc[i][j] = v8f{};

    for (int k0 = 0; k0 < EMB; k0 += 32) {
        v16bf af[4];
#pragma unroll
        for (int mt = 0; mt < 4; ++mt)
            af[mt] = load_a_bf16(x, EMB, m0w + mt * 16, k0, lane);
#pragma unroll
        for (int nt = 0; nt < 4; ++nt) {
            v16bf bfrag = load_b_bf16(w, EMB, k0, n0w + nt * 16, lane);
#pragma unroll
            for (int mt = 0; mt < 4; ++mt)
                acc[mt][nt] = wmma_bf16(af[mt], bfrag, acc[mt][nt]);
        }
    }

    // D layout: lane l, elem r -> row = r + (l>=16)*8, col = l&15
    const int col  = lane & 15;
    const int roff = (lane >> 4) * 8;
#pragma unroll
    for (int nt = 0; nt < 4; ++nt) {
        int   j  = n0w + nt * 16 + col;       // global qkv column, grouped (h, d, c)
        float bj = bias[j];
        int h  = j / 192;
        int rr = j - h * 192;
        int d  = rr / 3;
        int c  = rr - d * 3;
#pragma unroll
        for (int mt = 0; mt < 4; ++mt) {
#pragma unroll
            for (int r = 0; r < 8; ++r) {
                int m  = m0w + mt * 16 + roff + r;
                int bb = m >> 11;             // batch
                int nq = m & (NSEQ - 1);      // seq pos
                float v = acc[mt][nt][r] + bj;
                size_t bh = (size_t)(bb * NHEAD + h);
                if (c == 0) {
                    Qo[(bh * NSEQ + nq) * HDIM + d] = f2bf(v * 0.125f); // fold 1/sqrt(64)
                } else if (c == 1) {
                    Ko[(bh * NSEQ + nq) * HDIM + d] = f2bf(v);
                } else {
                    VTo[(bh * HDIM + d) * NSEQ + nq] = f2bf(v);         // V transposed
                }
            }
        }
    }
}

// ---------------------------------------------------------------------------
// Kernel 2: attention. One (b,h, 16-query block) per 128-thread workgroup.
// S[16][2048] bf16 in LDS (64 KB): phase-3 P fragments need NO conversion.
// ---------------------------------------------------------------------------
__global__ __launch_bounds__(128) void attn_kernel(
    const u16* __restrict__ Q, const u16* __restrict__ K,
    const u16* __restrict__ VT, u16* __restrict__ AO)
{
    __shared__ u16   S[16 * NSEQ];   // bf16 scores, then bf16 exp(P) in place
    __shared__ float lsum[16];

    const int lane = threadIdx.x & 31;
    const int wave = threadIdx.x >> 5;
    const int bh   = blockIdx.y;             // b*16 + h
    const int q0   = blockIdx.x * 16;
    const int b    = bh >> 4;
    const int h    = bh & 15;

    const u16* Qb = Q  + (size_t)bh * NSEQ * HDIM;
    const u16* Kb = K  + (size_t)bh * NSEQ * HDIM;
    const u16* Vb = VT + (size_t)bh * HDIM * NSEQ;

    // Q A-fragments for d-steps 0..31 and 32..63 (loaded once, reused 128x)
    const v16bf aq0 = load_a_bf16(Qb, HDIM, q0, 0, lane);
    const v16bf aq1 = load_a_bf16(Qb, HDIM, q0, 32, lane);

    const int col  = lane & 15;
    const int roff = (lane >> 4) * 8;

    // Phase 1: S = (scaled Q) @ K^T, 128 column tiles split over 4 waves
    for (int ct = wave; ct < NSEQ / 16; ct += 4) {
        int n0 = ct * 16;
        v8f s = v8f{};
        s = wmma_bf16(aq0, load_b_bf16(Kb, HDIM, 0, n0, lane), s);
        s = wmma_bf16(aq1, load_b_bf16(Kb, HDIM, 32, n0, lane), s);
#pragma unroll
        for (int r = 0; r < 8; ++r)
            S[(roff + r) * NSEQ + n0 + col] = f2bf(s[r]);
    }
    __syncthreads();

    // Phase 2: rowwise softmax over bf16 scores, vectorized 16B LDS accesses.
    // 8 threads per row; each owns a contiguous 256-element chunk.
    {
        const int row = threadIdx.x >> 3;    // 16 rows, 8 threads each
        const int sub = threadIdx.x & 7;
        u16* Sr = &S[row * NSEQ + sub * 256];
        float mx = -3.0e38f;
        for (int i = 0; i < 256; i += 8) {
            u16x8 v = *(const u16x8*)(Sr + i);
#pragma unroll
            for (int j = 0; j < 8; ++j) mx = fmaxf(mx, bf2f(v[j]));
        }
        mx = fmaxf(mx, __shfl_xor(mx, 1));
        mx = fmaxf(mx, __shfl_xor(mx, 2));
        mx = fmaxf(mx, __shfl_xor(mx, 4));
        float sum = 0.0f;
        for (int i = 0; i < 256; i += 8) {
            u16x8 v = *(const u16x8*)(Sr + i);
            u16x8 e;
#pragma unroll
            for (int j = 0; j < 8; ++j) {
                float ev = __expf(bf2f(v[j]) - mx);
                e[j] = f2bf(ev);
                sum += ev;
            }
            *(u16x8*)(Sr + i) = e;           // same slots: no cross-thread clash
        }
        sum += __shfl_xor(sum, 1);
        sum += __shfl_xor(sum, 2);
        sum += __shfl_xor(sum, 4);
        if (sub == 0) lsum[row] = sum;
    }
    __syncthreads();

    // Phase 3: O = P @ V ; wave owns d-tile [wave*16, wave*16+16).
    // A fragments come straight from LDS as bf16 (ds_load_b128, no VALU).
    const int d0 = wave * 16;
    v8f o = v8f{};
    for (int k0 = 0; k0 < NSEQ; k0 += 32) {
        v16bf ap = load_a_bf16(&S[0], NSEQ, 0, k0, lane);   // P fragment from LDS
        v16bf bp = load_b_bf16(Vb, NSEQ, k0, d0, lane);     // V^T: contiguous
        o = wmma_bf16(ap, bp, o);
    }
    // normalize by row sum, write [b, n, h*64 + d] bf16
#pragma unroll
    for (int r = 0; r < 8; ++r) {
        int   rq = roff + r;
        float v  = o[r] * (1.0f / lsum[rq]);
        int   q  = q0 + rq;
        AO[((size_t)(b * NSEQ + q)) * EMB + h * HDIM + d0 + col] = f2bf(v);
    }
}

// ---------------------------------------------------------------------------
// Kernel 3: out = AO @ w_proj^T + b_proj (fp32 out). Same tiling as kernel 1.
// ---------------------------------------------------------------------------
__global__ __launch_bounds__(128) void proj_gemm(
    const u16* __restrict__ A, const u16* __restrict__ w,
    const float* __restrict__ bias, float* __restrict__ out)
{
    const int lane = threadIdx.x & 31;
    const int wave = threadIdx.x >> 5;
    const int m0w  = blockIdx.y * 128 + (wave >> 1) * 64;
    const int n0w  = blockIdx.x * 128 + (wave & 1) * 64;

    v8f acc[4][4];
#pragma unroll
    for (int i = 0; i < 4; ++i)
#pragma unroll
        for (int j = 0; j < 4; ++j) acc[i][j] = v8f{};

    for (int k0 = 0; k0 < EMB; k0 += 32) {
        v16bf af[4];
#pragma unroll
        for (int mt = 0; mt < 4; ++mt)
            af[mt] = load_a_bf16(A, EMB, m0w + mt * 16, k0, lane);
#pragma unroll
        for (int nt = 0; nt < 4; ++nt) {
            v16bf bfrag = load_b_bf16(w, EMB, k0, n0w + nt * 16, lane);
#pragma unroll
            for (int mt = 0; mt < 4; ++mt)
                acc[mt][nt] = wmma_bf16(af[mt], bfrag, acc[mt][nt]);
        }
    }

    const int col  = lane & 15;
    const int roff = (lane >> 4) * 8;
#pragma unroll
    for (int nt = 0; nt < 4; ++nt) {
        int   j  = n0w + nt * 16 + col;
        float bj = bias[j];
#pragma unroll
        for (int mt = 0; mt < 4; ++mt) {
#pragma unroll
            for (int r = 0; r < 8; ++r) {
                int m = m0w + mt * 16 + roff + r;
                out[(size_t)m * EMB + j] = acc[mt][nt][r] + bj;
            }
        }
    }
}

// ---------------------------------------------------------------------------
extern "C" void kernel_launch(void* const* d_in, const int* in_sizes, int n_in,
                              void* d_out, int out_size, void* d_ws, size_t ws_size,
                              hipStream_t stream)
{
    (void)in_sizes; (void)n_in; (void)out_size; (void)ws_size;
    const float* x      = (const float*)d_in[0];
    const float* w_qkv  = (const float*)d_in[1];
    const float* b_qkv  = (const float*)d_in[2];
    const float* w_proj = (const float*)d_in[3];
    const float* b_proj = (const float*)d_in[4];
    float* out = (float*)d_out;

    const size_t SZ   = (size_t)MTOT * EMB;          // 8388608
    const size_t WQSZ = (size_t)NQKV * EMB;          // 3145728
    const size_t WPSZ = (size_t)EMB * EMB;           // 1048576

    u16* Q    = (u16*)d_ws;
    u16* K    = Q + SZ;
    u16* VT   = K + SZ;
    u16* AO   = VT + SZ;
    u16* Xbf  = AO + SZ;
    u16* Wqkv = Xbf + SZ;
    u16* Wprj = Wqkv + WQSZ;                          // total ws use: ~88 MiB

    // pre-pass conversions (counts all divisible by 8*256)
    cvt_f32_bf16<<<dim3((unsigned)(SZ   / 2048)), dim3(256), 0, stream>>>(x,      Xbf);
    cvt_f32_bf16<<<dim3((unsigned)(WQSZ / 2048)), dim3(256), 0, stream>>>(w_qkv,  Wqkv);
    cvt_f32_bf16<<<dim3((unsigned)(WPSZ / 2048)), dim3(256), 0, stream>>>(w_proj, Wprj);

    qkv_gemm_scatter<<<dim3(NQKV / 128, MTOT / 128), dim3(128), 0, stream>>>(
        Xbf, Wqkv, b_qkv, Q, K, VT);
    attn_kernel<<<dim3(NSEQ / 16, NBATCH * NHEAD), dim3(128), 0, stream>>>(
        Q, K, VT, AO);
    proj_gemm<<<dim3(EMB / 128, MTOT / 128), dim3(128), 0, stream>>>(
        AO, Wprj, b_proj, out);
}
